// ModelB_30562987278954
// MI455X (gfx1250) — matrix operand order repaired and checked by reading, not compiled, not run on hardware
//
#include <hip/hip_runtime.h>
#include <cmath>

// ---------------------------------------------------------------------------
// CDNA5 (gfx1250) wave32 WMMA types
// ---------------------------------------------------------------------------
typedef _Float16 h16 __attribute__((ext_vector_type(16)));
typedef _Float16 h8  __attribute__((ext_vector_type(8)));
typedef float    f8  __attribute__((ext_vector_type(8)));
typedef float    fv4 __attribute__((ext_vector_type(4)));

__device__ __forceinline__ f8 wmma_f32_16x16x32_f16(h16 a, h16 b, f8 c) {
  return __builtin_amdgcn_wmma_f32_16x16x32_f16(false, a, false, b, (short)0, c,
                                                false, false);
}

__device__ __forceinline__ h16 join16(h8 lo, h8 hi) {
  return __builtin_shufflevector(lo, hi, 0, 1, 2, 3, 4, 5, 6, 7, 8, 9, 10, 11,
                                 12, 13, 14, 15);
}

// f32 -> f16 fragment half from 4 float4s (for external weights kept in f32)
__device__ __forceinline__ void cvt16(h16& d, fv4 a0, fv4 a1, fv4 a2, fv4 a3) {
#pragma unroll
  for (int e = 0; e < 4; ++e) {
    d[e]      = (_Float16)a0[e];
    d[e + 4]  = (_Float16)a1[e];
    d[e + 8]  = (_Float16)a2[e];
    d[e + 12] = (_Float16)a3[e];
  }
}

// ===========================================================================
// GEMM: out[m,n] = act( A(f16)[MxK] @ W(f32)[KxN] + bias[n] )
// REQUIRES: lda%8==0, ldw%4==0, N%16==0 (HAS1: N%32==0), K%32==0, aligned A/W.
// One wave computes a 16(M) x 32(N) tile (2 accumulators, A reused).
// Out-of-range M rows are clamped (their outputs are never stored).
// ===========================================================================
template <bool HAS1, bool STORE_HALF>
__global__ __launch_bounds__(32)
void gemm_bias_act_kernel(const _Float16* __restrict__ A, int lda,
                          const float* __restrict__ W, int ldw,
                          const float* __restrict__ bias,
                          void* __restrict__ outp, int ldo,
                          int M, int N, int K, int relu) {
  const int lane = threadIdx.x;
  const int r = lane & 15, hi = lane >> 4;
  const int m0 = blockIdx.x * 16;
  const int n0 = blockIdx.y * 32;
  const int n1 = n0 + 16;
  const int m = m0 + r;
  const int mc = (m < M) ? m : (M - 1);
  const _Float16* arow = A + (size_t)mc * lda;
  f8 acc0 = {}, acc1 = {};
#pragma unroll 2
  for (int k0 = 0; k0 < K; k0 += 32) {
    const h8 alo = *(const h8*)(arow + k0 + hi * 8);
    const h8 ahi = *(const h8*)(arow + k0 + 16 + hi * 8);
    const h16 a = join16(alo, ahi);
    const float* wrow = W + (size_t)(k0 + lane) * ldw;
    __builtin_prefetch((const void*)(wrow + 32 * (size_t)ldw), 0, 1);
    h16 b0;
#pragma unroll
    for (int s = 0; s < 16; ++s) {
      const int kk = (s < 8) ? (hi * 8 + s) : (16 + hi * 8 + (s - 8));
      b0[s] = (_Float16)W[(size_t)(k0 + kk) * ldw + n0 + r];
    }
    acc0 = wmma_f32_16x16x32_f16(a, b0, acc0);
    if (HAS1) {
      h16 b1;
#pragma unroll
      for (int s = 0; s < 16; ++s) {
        const int kk = (s < 8) ? (hi * 8 + s) : (16 + hi * 8 + (s - 8));
        b1[s] = (_Float16)W[(size_t)(k0 + kk) * ldw + n1 + r];
      }
      acc1 = wmma_f32_16x16x32_f16(a, b1, acc1);
    }
  }
  const int nc = lane & 15;
#pragma unroll
  for (int rr = 0; rr < 8; ++rr) {
    int mm = m0 + rr + 8 * hi;
    if (mm < M) {
      float v0 = acc0[rr] + bias[n0 + nc];
      if (relu) v0 = fmaxf(v0, 0.f);
      if (STORE_HALF)
        ((_Float16*)outp)[(size_t)mm * ldo + n0 + nc] = (_Float16)v0;
      else
        ((float*)outp)[(size_t)mm * ldo + n0 + nc] = v0;
      if (HAS1) {
        float v1 = acc1[rr] + bias[n1 + nc];
        if (relu) v1 = fmaxf(v1, 0.f);
        if (STORE_HALF)
          ((_Float16*)outp)[(size_t)mm * ldo + n1 + nc] = (_Float16)v1;
        else
          ((float*)outp)[(size_t)mm * ldo + n1 + nc] = v1;
      }
    }
  }
}

// ===========================================================================
// Layout prep: featT[b][p][c] (f16 pixel-major), w3T[tap][ci][co] (f16),
// wt[k][80] (f16) + bt[80], wh[k][16] (f32, cls|reg fused) + bh[16]
// ===========================================================================
__global__ __launch_bounds__(256)
void transpose_feat_kernel(const float* __restrict__ feat, int HW, int total,
                           _Float16* __restrict__ featT) {
  int id = blockIdx.x * 256 + threadIdx.x;
  if (id >= total) return;
  int p = id % HW;
  int rest = id / HW;
  int c = rest & 255;
  int b = rest >> 8;
  featT[((size_t)b * HW + p) * 256 + c] = (_Float16)feat[id];
}

__global__ __launch_bounds__(256)
void pack_weights_kernel(const float* __restrict__ w3, const float* __restrict__ wc,
                         const float* __restrict__ bc, const float* __restrict__ wr,
                         const float* __restrict__ br, const float* __restrict__ cw,
                         const float* __restrict__ cb, const float* __restrict__ rw,
                         const float* __restrict__ rb,
                         _Float16* __restrict__ w3T, _Float16* __restrict__ wt,
                         float* __restrict__ bt, float* __restrict__ wh,
                         float* __restrict__ bh) {
  const int NW3 = 9 * 256 * 256;
  int id = blockIdx.x * 256 + threadIdx.x;
  if (id < NW3) {
    int co = id & 255;
    int rest = id >> 8;
    int ci = rest & 255;
    int tap = rest >> 8;
    int ky = tap / 3, kx = tap % 3;
    w3T[id] = (_Float16)w3[(((size_t)co * 256 + ci) * 3 + ky) * 3 + kx];
  } else if (id < NW3 + 256 * 80) {
    int t = id - NW3;
    int k = t / 80, j = t % 80;
    float v = 0.f;
    if (j < 15) v = wc[(size_t)j * 256 + k];
    else if (j < 75) v = wr[(size_t)(j - 15) * 256 + k];
    wt[t] = (_Float16)v;
  } else if (id < NW3 + 256 * 80 + 80) {
    int j = id - NW3 - 256 * 80;
    bt[j] = (j < 15) ? bc[j] : ((j < 75) ? br[j - 15] : 0.f);
  } else if (id < NW3 + 256 * 80 + 80 + 1024 * 16) {
    int t = id - (NW3 + 256 * 80 + 80);
    int k = t / 16, j = t % 16;
    float v = 0.f;
    if (j < 3) v = cw[(size_t)k * 3 + j];
    else if (j < 15) v = rw[(size_t)k * 12 + (j - 3)];
    wh[t] = v;
  } else if (id < NW3 + 256 * 80 + 80 + 1024 * 16 + 16) {
    int j = id - (NW3 + 256 * 80 + 80 + 1024 * 16);
    bh[j] = (j < 3) ? cb[j] : ((j < 15) ? rb[j - 3] : 0.f);
  }
}

// ===========================================================================
// RPN 3x3 conv (pad=1) + bias + ReLU, implicit GEMM, all-f16 operands.
// A: featT[b][pix][ci] (halo via clamp + 0/1 f16 mask), B: w3T[tap][ci][co].
// Writes tT[b][p][co] (f16).
// ===========================================================================
__global__ __launch_bounds__(32)
void rpn_conv3x3_kernel(const _Float16* __restrict__ featT, int H, int W,
                        const _Float16* __restrict__ w3T,
                        const float* __restrict__ bias,
                        _Float16* __restrict__ tT) {
  const int lane = threadIdx.x;
  const int r = lane & 15, hi = lane >> 4;
  const int bb = blockIdx.z;
  const int HW = H * W;
  const int m0 = blockIdx.x * 16, n0 = blockIdx.y * 16;
  const int p = m0 + r;
  const int pc = min(p, HW - 1);
  const int y = pc / W, x = pc % W;
  f8 acc = {};
  for (int tap = 0; tap < 9; ++tap) {
    const int ky = tap / 3, kx = tap % 3;
    const int yy = y + ky - 1, xx = x + kx - 1;
    const _Float16 mk =
        (p < HW && yy >= 0 && yy < H && xx >= 0 && xx < W) ? (_Float16)1.f
                                                           : (_Float16)0.f;
    const int yc = min(max(yy, 0), H - 1);
    const int xc = min(max(xx, 0), W - 1);
    const _Float16* src =
        featT + ((size_t)bb * HW + (size_t)(yc * W + xc)) * 256;
    const _Float16* wbase = w3T + (size_t)tap * 256 * 256 + n0;
#pragma unroll 2
    for (int c0 = 0; c0 < 256; c0 += 32) {
      h8 alo = *(const h8*)(src + c0 + hi * 8);
      h8 ahi = *(const h8*)(src + c0 + 16 + hi * 8);
      alo = alo * mk;
      ahi = ahi * mk;
      const h16 a = join16(alo, ahi);
      const _Float16* wcol = wbase + (size_t)c0 * 256 + r;
      h16 b;
#pragma unroll
      for (int s = 0; s < 16; ++s) {
        const int kk = (s < 8) ? (hi * 8 + s) : (16 + hi * 8 + (s - 8));
        b[s] = wcol[(size_t)kk * 256];
      }
      acc = wmma_f32_16x16x32_f16(a, b, acc);
    }
  }
  const int co = n0 + (lane & 15);
#pragma unroll
  for (int rr = 0; rr < 8; ++rr) {
    int pp = m0 + rr + 8 * hi;
    if (pp < HW) {
      float v = fmaxf(acc[rr] + bias[co], 0.f);
      tT[((size_t)bb * HW + pp) * 256 + co] = (_Float16)v;
    }
  }
}

// ===========================================================================
// RPN 1x1 heads: [HW x 256](f16) @ wt[256 x 80](f16)
// cols 0..14 -> logits[b][p*15+a], 15..74 -> deltas[b][p*60 + (a*4+c)]
// ===========================================================================
__global__ __launch_bounds__(32)
void rpn_heads_kernel(const _Float16* __restrict__ tT, int HW,
                      const _Float16* __restrict__ wt,
                      const float* __restrict__ bt,
                      float* __restrict__ logits, float* __restrict__ deltas) {
  const int lane = threadIdx.x;
  const int r = lane & 15, hi = lane >> 4;
  const int bb = blockIdx.z;
  const int m0 = blockIdx.x * 16, n0 = blockIdx.y * 16;
  const int p = m0 + r;
  const int pc = min(p, HW - 1);
  const _Float16* arow = tT + ((size_t)bb * HW + pc) * 256;
  f8 acc = {};
#pragma unroll 2
  for (int k0 = 0; k0 < 256; k0 += 32) {
    const h16 a = join16(*(const h8*)(arow + k0 + hi * 8),
                         *(const h8*)(arow + k0 + 16 + hi * 8));
    const _Float16* wcol = wt + (size_t)k0 * 80 + n0 + r;
    h16 b;
#pragma unroll
    for (int s = 0; s < 16; ++s) {
      const int kk = (s < 8) ? (hi * 8 + s) : (16 + hi * 8 + (s - 8));
      b[s] = wcol[(size_t)kk * 80];
    }
    acc = wmma_f32_16x16x32_f16(a, b, acc);
  }
  const int j = n0 + (lane & 15);
#pragma unroll
  for (int rr = 0; rr < 8; ++rr) {
    int pp = m0 + rr + 8 * hi;
    if (pp < HW) {
      if (j < 15)
        logits[(size_t)bb * HW * 15 + (size_t)pp * 15 + j] = acc[rr] + bt[j];
      else if (j < 75)
        deltas[(size_t)bb * HW * 60 + (size_t)pp * 60 + (j - 15)] =
            acc[rr] + bt[j];
    }
  }
}

// ===========================================================================
// Split fused head output ho[200x16] -> cls[200x3] | reg[200x12] in d_out
// ===========================================================================
__global__ __launch_bounds__(256)
void split_heads_kernel(const float* __restrict__ ho, float* __restrict__ out) {
  int id = blockIdx.x * 256 + threadIdx.x;
  if (id >= 200 * 15) return;
  int m = id / 15, j = id % 15;
  float v = ho[m * 16 + j];
  if (j < 3) out[m * 3 + j] = v;
  else out[600 + m * 12 + (j - 3)] = v;
}

// ===========================================================================
// Top-500 (stable: ties -> smaller index first), one block per batch image.
// ===========================================================================
__global__ __launch_bounds__(256)
void topk_kernel(const float* __restrict__ scores, int N,
                 float* __restrict__ topv, int* __restrict__ topi) {
  const int b = blockIdx.x;
  const float* s = scores + (size_t)b * N;
  float* tv = topv + b * 500;
  int* ti = topi + b * 500;
  __shared__ float sv[256];
  __shared__ int si[256];
  __shared__ float pvs;
  __shared__ int pis;
  if (threadIdx.x == 0) { pvs = 3.4e38f; pis = -1; }
  __syncthreads();
  for (int it = 0; it < 500; ++it) {
    const float pv = pvs;
    const int pi = pis;
    float bestv = -3.4e38f;
    int besti = 0x7fffffff;
    for (int i = threadIdx.x; i < N; i += 256) {
      float v = s[i];
      bool elig = (v < pv) || (v == pv && i > pi);
      if (elig && (v > bestv || (v == bestv && i < besti))) { bestv = v; besti = i; }
    }
    sv[threadIdx.x] = bestv;
    si[threadIdx.x] = besti;
    __syncthreads();
    for (int o = 128; o > 0; o >>= 1) {
      if (threadIdx.x < o) {
        float v2 = sv[threadIdx.x + o];
        int i2 = si[threadIdx.x + o];
        if (v2 > sv[threadIdx.x] ||
            (v2 == sv[threadIdx.x] && i2 < si[threadIdx.x])) {
          sv[threadIdx.x] = v2;
          si[threadIdx.x] = i2;
        }
      }
      __syncthreads();
    }
    if (threadIdx.x == 0) { tv[it] = sv[0]; ti[it] = si[0]; pvs = sv[0]; pis = si[0]; }
    __syncthreads();
  }
}

// ===========================================================================
// Decode + clip + sigmoid + level offset.
// ===========================================================================
__global__ __launch_bounds__(256)
void decode_kernel(const float* __restrict__ d0, const float* __restrict__ d1,
                   const float* __restrict__ d2, const float* __restrict__ topv,
                   const int* __restrict__ topi, float* __restrict__ boxes,
                   float* __restrict__ ioub, float* __restrict__ scores) {
  int id = blockIdx.x * blockDim.x + threadIdx.x;
  if (id >= 2 * 1500) return;
  const int b = id / 1500, j = id % 1500;
  const int l = j / 500, i = j % 500;
  int Wl, strd, HWl;
  const float* dl;
  if (l == 0)      { Wl = 50; strd = 16; HWl = 2500; dl = d0; }
  else if (l == 1) { Wl = 25; strd = 32; HWl = 625;  dl = d1; }
  else             { Wl = 13; strd = 64; HWl = 169;  dl = d2; }
  const float v = topv[(l * 2 + b) * 500 + i];
  const int n = topi[(l * 2 + b) * 500 + i];
  const int a = n % 15, cell = n / 15;
  const int x = cell % Wl, y = cell / Wl;
  const float SZ[5] = {32.f, 64.f, 128.f, 256.f, 512.f};
  const float SQ[3] = {0.70710678118654752f, 1.f, 1.41421356237309505f};
  const float s = SZ[a / 3], q = SQ[a % 3];
  const float ha = s * q, wa = s / q;
  const float cxa = (float)x * (float)strd, cya = (float)y * (float)strd;
  const float* dd = dl + (size_t)b * HWl * 60 + (size_t)n * 4;
  const float BBOX_CLAMP = 4.1351665567423560f; // log(1000/16)
  float dx = dd[0], dy = dd[1];
  float dw = fminf(dd[2], BBOX_CLAMP), dh = fminf(dd[3], BBOX_CLAMP);
  float cx = dx * wa + cxa, cy = dy * ha + cya;
  float w = expf(dw) * wa, h = expf(dh) * ha;
  float x1 = fminf(fmaxf(cx - 0.5f * w, 0.f), 800.f);
  float y1 = fminf(fmaxf(cy - 0.5f * h, 0.f), 800.f);
  float x2 = fminf(fmaxf(cx + 0.5f * w, 0.f), 800.f);
  float y2 = fminf(fmaxf(cy + 0.5f * h, 0.f), 800.f);
  float sc = 1.f / (1.f + expf(-v));
  if ((x2 - x1 < 0.001f) || (y2 - y1 < 0.001f)) sc = -1.f;
  const float offv = (float)l * 900.f;
  const size_t base = (size_t)b * 6000 + (size_t)j * 4;
  boxes[base + 0] = x1; boxes[base + 1] = y1; boxes[base + 2] = x2; boxes[base + 3] = y2;
  ioub[base + 0] = x1 + offv; ioub[base + 1] = y1 + offv;
  ioub[base + 2] = x2 + offv; ioub[base + 3] = y2 + offv;
  scores[b * 1500 + j] = sc;
}

// ===========================================================================
// NMS: bitonic sort 2048 in LDS (desc score, asc idx), sequential suppression.
// ===========================================================================
#define NMS_N 1500
#define NMS_PAD 2048
__global__ __launch_bounds__(256)
void nms_kernel(const float* __restrict__ boxes, const float* __restrict__ ioub,
                const float* __restrict__ scores, float* __restrict__ props,
                float* __restrict__ pscores) {
  __shared__ float ss[NMS_PAD];
  __shared__ int so[NMS_PAD];
  __shared__ float bi[NMS_N * 4];
  __shared__ int keep[NMS_N];
  const int b = blockIdx.x, tid = threadIdx.x, nt = blockDim.x;
  for (int i = tid; i < NMS_PAD; i += nt) {
    ss[i] = (i < NMS_N) ? scores[b * NMS_N + i] : -1e30f;
    so[i] = i;
  }
  __syncthreads();
  for (int k = 2; k <= NMS_PAD; k <<= 1) {
    for (int j = k >> 1; j > 0; j >>= 1) {
      for (int i = tid; i < NMS_PAD; i += nt) {
        int ixj = i ^ j;
        if (ixj > i) {
          float va = ss[i], vb = ss[ixj];
          int ia = so[i], ib = so[ixj];
          bool up = ((i & k) == 0);
          bool bef_ba = (vb > va) || (vb == va && ib < ia);
          bool bef_ab = (va > vb) || (va == vb && ia < ib);
          if (up ? bef_ba : bef_ab) {
            ss[i] = vb; so[i] = ib; ss[ixj] = va; so[ixj] = ia;
          }
        }
      }
      __syncthreads();
    }
  }
  for (int i = tid; i < NMS_N; i += nt) {
    int o = so[i];
    bi[i * 4 + 0] = ioub[b * NMS_N * 4 + o * 4 + 0];
    bi[i * 4 + 1] = ioub[b * NMS_N * 4 + o * 4 + 1];
    bi[i * 4 + 2] = ioub[b * NMS_N * 4 + o * 4 + 2];
    bi[i * 4 + 3] = ioub[b * NMS_N * 4 + o * 4 + 3];
    keep[i] = 1;
  }
  __syncthreads();
  for (int i = 0; i < NMS_N - 1; ++i) {
    if (keep[i]) {
      float ax1 = bi[i * 4], ay1 = bi[i * 4 + 1], ax2 = bi[i * 4 + 2],
            ay2 = bi[i * 4 + 3];
      float aarea = (ax2 - ax1) * (ay2 - ay1);
      for (int j = i + 1 + tid; j < NMS_N; j += nt) {
        float bx1 = bi[j * 4], by1 = bi[j * 4 + 1], bx2 = bi[j * 4 + 2],
              by2 = bi[j * 4 + 3];
        float barea = (bx2 - bx1) * (by2 - by1);
        float iw = fmaxf(fminf(ax2, bx2) - fmaxf(ax1, bx1), 0.f);
        float ih = fmaxf(fminf(ay2, by2) - fmaxf(ay1, by1), 0.f);
        float inter = iw * ih;
        float iou = inter / (aarea + barea - inter + 1e-6f);
        if (iou > 0.7f) keep[j] = 0;
      }
    }
    __syncthreads();
  }
  if (tid == 0) {
    int r = 0;
    for (int i = 0; i < NMS_N && r < 100; ++i) {
      if (keep[i]) {
        int o = so[i];
        props[b * 400 + r * 4 + 0] = boxes[b * 6000 + o * 4 + 0];
        props[b * 400 + r * 4 + 1] = boxes[b * 6000 + o * 4 + 1];
        props[b * 400 + r * 4 + 2] = boxes[b * 6000 + o * 4 + 2];
        props[b * 400 + r * 4 + 3] = boxes[b * 6000 + o * 4 + 3];
        pscores[b * 100 + r] = ss[i];
        ++r;
      }
    }
    for (; r < 100; ++r) {
      props[b * 400 + r * 4 + 0] = 0.f;
      props[b * 400 + r * 4 + 1] = 0.f;
      props[b * 400 + r * 4 + 2] = 0.f;
      props[b * 400 + r * 4 + 3] = 0.f;
      pscores[b * 100 + r] = -1.f;
    }
  }
}

// ===========================================================================
// RoIAlign 7x7 (SR=2): one block per roi, one lane per channel. f16 output.
// ===========================================================================
__global__ __launch_bounds__(256)
void roi_align_kernel(const float* __restrict__ f0, const float* __restrict__ f1,
                      const float* __restrict__ f2, const float* __restrict__ rois,
                      _Float16* __restrict__ pooled) {
  const int n = blockIdx.x;
  const int c = threadIdx.x;
  const int b = n / 100;
  const float x1 = rois[n * 4 + 0], y1 = rois[n * 4 + 1];
  const float x2 = rois[n * 4 + 2], y2 = rois[n * 4 + 3];
  const float area = (x2 - x1) * (y2 - y1);
  float lvl = floorf(4.f + log2f(sqrtf(area) / 224.f + 1e-8f));
  lvl = fminf(fmaxf(lvl, 4.f), 6.f);
  const int li = (int)lvl - 4;
  const float* f;
  int H, W;
  float scale;
  if (li == 0)      { f = f0; H = 50; W = 50; scale = 1.f / 16.f; }
  else if (li == 1) { f = f1; H = 25; W = 25; scale = 1.f / 32.f; }
  else              { f = f2; H = 13; W = 13; scale = 1.f / 64.f; }
  const float* img = f + ((size_t)b * 256 + c) * H * W;
  const float sx1 = x1 * scale, sy1 = y1 * scale;
  const float rw = fmaxf(x2 * scale - sx1, 1.f);
  const float rh = fmaxf(y2 * scale - sy1, 1.f);
  for (int py = 0; py < 7; ++py) {
    for (int px = 0; px < 7; ++px) {
      float acc = 0.f;
      for (int sy = 0; sy < 2; ++sy) {
        for (int sx = 0; sx < 2; ++sx) {
          float ox = ((float)(px * 2 + sx) + 0.5f) / 14.f;
          float oy = ((float)(py * 2 + sy) + 0.5f) / 14.f;
          float X = fminf(fmaxf(sx1 + ox * rw, 0.f), (float)(W - 1));
          float Y = fminf(fmaxf(sy1 + oy * rh, 0.f), (float)(H - 1));
          int x0 = (int)floorf(X), y0 = (int)floorf(Y);
          int xp = min(x0 + 1, W - 1), yp = min(y0 + 1, H - 1);
          float lx = X - (float)x0, ly = Y - (float)y0;
          float v00 = img[y0 * W + x0], v01 = img[y0 * W + xp];
          float v10 = img[yp * W + x0], v11 = img[yp * W + xp];
          acc += v00 * (1.f - ly) * (1.f - lx) + v01 * (1.f - ly) * lx +
                 v10 * ly * (1.f - lx) + v11 * ly * lx;
        }
      }
      pooled[(size_t)n * 12544 + (size_t)c * 49 + py * 7 + px] =
          (_Float16)(acc * 0.25f);
    }
  }
}

// ===========================================================================
// Host-side launch
// ===========================================================================
extern "C" void kernel_launch(void* const* d_in, const int* in_sizes, int n_in,
                              void* d_out, int out_size, void* d_ws,
                              size_t ws_size, hipStream_t stream) {
  const float* feat[3] = {(const float*)d_in[0], (const float*)d_in[1],
                          (const float*)d_in[2]};
  const float* rpn_conv_w = (const float*)d_in[3];
  const float* rpn_conv_b = (const float*)d_in[4];
  const float* rpn_cls_w = (const float*)d_in[5];
  const float* rpn_cls_b = (const float*)d_in[6];
  const float* rpn_reg_w = (const float*)d_in[7];
  const float* rpn_reg_b = (const float*)d_in[8];
  const float* fc1_w = (const float*)d_in[9];
  const float* fc1_b = (const float*)d_in[10];
  const float* fc2_w = (const float*)d_in[11];
  const float* fc2_b = (const float*)d_in[12];
  const float* cls_w = (const float*)d_in[13];
  const float* cls_b = (const float*)d_in[14];
  const float* reg_w = (const float*)d_in[15];
  const float* reg_b = (const float*)d_in[16];
  float* out = (float*)d_out;  // [600 cls | 2400 reg | 800 props | 200 pscores]

  const int Hs[3] = {50, 25, 13};
  const int HWs[3] = {2500, 625, 169};
  const int B = 2;

  char* wsb = (char*)d_ws;
  size_t off = 0;
  auto alloc = [&](size_t bytes) {
    void* p = wsb + off;
    off += (bytes + 15) & ~(size_t)15;
    return p;
  };
  _Float16* featT[3];
  _Float16* tT[3];
  float* lg[3];
  float* de[3];
  for (int l = 0; l < 3; ++l)
    featT[l] = (_Float16*)alloc((size_t)B * 256 * HWs[l] * 2);
  for (int l = 0; l < 3; ++l)
    tT[l] = (_Float16*)alloc((size_t)B * 256 * HWs[l] * 2);
  for (int l = 0; l < 3; ++l) lg[l] = (float*)alloc((size_t)B * HWs[l] * 15 * 4);
  for (int l = 0; l < 3; ++l) de[l] = (float*)alloc((size_t)B * HWs[l] * 60 * 4);
  _Float16* w3T = (_Float16*)alloc((size_t)9 * 256 * 256 * 2);
  _Float16* wt = (_Float16*)alloc((size_t)256 * 80 * 2);
  float* bt = (float*)alloc(80 * 4);
  float* wh = (float*)alloc((size_t)1024 * 16 * 4);
  float* bh = (float*)alloc(16 * 4);
  float* topv = (float*)alloc(3 * B * 500 * 4);
  int* topi = (int*)alloc(3 * B * 500 * 4);
  float* boxes = (float*)alloc((size_t)B * 1500 * 4 * 4);
  float* ioubp = (float*)alloc((size_t)B * 1500 * 4 * 4);
  float* scoresp = (float*)alloc((size_t)B * 1500 * 4);
  _Float16* pooled = (_Float16*)alloc((size_t)200 * 12544 * 2);
  _Float16* h1 = (_Float16*)alloc((size_t)200 * 1024 * 2);
  _Float16* h2 = (_Float16*)alloc((size_t)200 * 1024 * 2);
  float* ho = (float*)alloc((size_t)200 * 16 * 4);
  (void)in_sizes; (void)n_in; (void)out_size; (void)ws_size;

  // --- layout prep
  {
    const int total = 9 * 256 * 256 + 256 * 80 + 80 + 1024 * 16 + 16;
    pack_weights_kernel<<<(total + 255) / 256, 256, 0, stream>>>(
        rpn_conv_w, rpn_cls_w, rpn_cls_b, rpn_reg_w, rpn_reg_b, cls_w, cls_b,
        reg_w, reg_b, w3T, wt, bt, wh, bh);
  }
  for (int l = 0; l < 3; ++l) {
    int total = B * 256 * HWs[l];
    transpose_feat_kernel<<<(total + 255) / 256, 256, 0, stream>>>(
        feat[l], HWs[l], total, featT[l]);
  }

  // --- RPN per level: 3x3 conv (WMMA implicit GEMM), 1x1 heads (WMMA), top-k
  for (int l = 0; l < 3; ++l) {
    const int H = Hs[l], W = Hs[l], HW = HWs[l];
    dim3 gc((HW + 15) / 16, 16, B);
    rpn_conv3x3_kernel<<<gc, 32, 0, stream>>>(featT[l], H, W, w3T, rpn_conv_b,
                                              tT[l]);
    dim3 gh((HW + 15) / 16, 5, B);
    rpn_heads_kernel<<<gh, 32, 0, stream>>>(tT[l], HW, wt, bt, lg[l], de[l]);
    topk_kernel<<<B, 256, 0, stream>>>(lg[l], HW * 15, topv + l * B * 500,
                                       topi + l * B * 500);
  }

  // --- decode + NMS -> proposals straight into d_out
  decode_kernel<<<(B * 1500 + 255) / 256, 256, 0, stream>>>(
      de[0], de[1], de[2], topv, topi, boxes, ioubp, scoresp);
  nms_kernel<<<B, 256, 0, stream>>>(boxes, ioubp, scoresp, out + 3000,
                                    out + 3800);

  // --- RoIAlign (f16 pooled features, row-major for FC1)
  roi_align_kernel<<<200, 256, 0, stream>>>(feat[0], feat[1], feat[2],
                                            out + 3000, pooled);

  // --- FC chain (WMMA GEMMs, 16x32 tiles, f16 activations)
  gemm_bias_act_kernel<true, true><<<dim3(13, 32), 32, 0, stream>>>(
      pooled, 12544, fc1_w, 1024, fc1_b, h1, 1024, 200, 1024, 12544, 1);
  gemm_bias_act_kernel<true, true><<<dim3(13, 32), 32, 0, stream>>>(
      h1, 1024, fc2_w, 1024, fc2_b, h2, 1024, 200, 1024, 1024, 1);
  gemm_bias_act_kernel<false, false><<<dim3(13, 1), 32, 0, stream>>>(
      h2, 1024, wh, 16, bh, ho, 16, 200, 16, 1024, 0);
  split_heads_kernel<<<(200 * 15 + 255) / 256, 256, 0, stream>>>(ho, out);
}
